// GraphConv_6597069766680
// MI455X (gfx1250) — matrix operand order repaired and checked by reading, not compiled, not run on hardware
//
#include <hip/hip_runtime.h>

typedef __bf16 bf16_t;
typedef __attribute__((ext_vector_type(16))) __bf16 bf16x16;
typedef __attribute__((ext_vector_type(8)))  __bf16 bf16x8;
typedef __attribute__((ext_vector_type(4)))  __bf16 bf16x4;
typedef __attribute__((ext_vector_type(8)))  float  f32x8;
typedef __attribute__((ext_vector_type(4)))  float  f32x4;
typedef __attribute__((ext_vector_type(4)))  unsigned int u32x4;
typedef __attribute__((ext_vector_type(8)))  unsigned int u32x8;

#define NROWS 8192
#define INF   512
#define OUTF  512

// ---------------------------------------------------------------------------
// TDM: DMA a [512 rows x 32 cols] bf16 tile of hT (row stride 8192 elements)
// into LDS, with hardware padding: 16 dwords data + 4 dwords pad per row
// -> 80-byte LDS row pitch (conflict-free b128 fragment reads).
// D# built per CDNA5 ISA sec.8; 2D tile so descriptor groups 2/3 are NULL.
// ---------------------------------------------------------------------------
__device__ __forceinline__ void tdm_load_b_tile(unsigned lds_byte_off,
                                                const bf16_t* gptr) {
    unsigned long long ga = (unsigned long long)gptr;
    u32x4 g0;
    g0[0] = 1u;                                   // count=1, user mode, no gather
    g0[1] = lds_byte_off;                         // lds_addr
    g0[2] = (unsigned)(ga & 0xffffffffu);         // global_addr[31:0]
    g0[3] = (unsigned)((ga >> 32) & 0x01ffffffu)  // global_addr[56:32]
          | (2u << 30);                           // type = 2 ("image")
    u32x8 g1;
    g1[0] = (1u << 16)                            // data_size = 1 -> 2 bytes
          | (1u << 20)                            // pad_enable
          | (3u << 22)                            // pad_interval: 16 dwords
          | (3u << 25);                           // pad_amount:   4 dwords
    g1[1] = (NROWS & 0xffffu) << 16;              // tensor_dim0[15:0] = 8192
    g1[2] = (OUTF  & 0xffffu) << 16;              // dim0[31:16]=0 | tensor_dim1 = 512
    g1[3] = (32u << 16);                          // dim1[31:16]=0 | tile_dim0 = 32
    g1[4] = 512u;                                 // tile_dim1 = 512, tile_dim2 = 0
    g1[5] = (unsigned)NROWS;                      // tensor_dim0_stride[31:0] = 8192
    g1[6] = 0u;                                   // stride hi | dim1_stride lo
    g1[7] = 0u;
    asm volatile("tensor_load_to_lds %0, %1" :: "s"(g0), "s"(g1) : "memory");
}

// ---------------------------------------------------------------------------
// Kernel 1: hT[o][n] = sum_k input[n][k] * W[o][k] + b[o]   (bf16 output)
// One 16x16 output tile per wave, K swept 32 at a time with WMMA bf16.
// ---------------------------------------------------------------------------
__global__ __launch_bounds__(256)
void linear_bf16_kernel(const float* __restrict__ input,  // [NROWS][INF]
                        const float* __restrict__ W,      // [OUTF][INF]
                        const float* __restrict__ bias,   // [OUTF]
                        bf16_t* __restrict__ hT)          // [OUTF][NROWS]
{
    const int wave = threadIdx.x >> 5;
    const int lane = threadIdx.x & 31;
    const int l16  = lane & 15;
    const int hi   = lane >> 4;

    const int tile  = blockIdx.x * 8 + wave;   // 16384 tiles total
    const int mTile = tile >> 5;               // rows of input, 16 each
    const int oTile = tile & 31;               // output features, 16 each
    const int mBase = mTile * 16;
    const int oBase = oTile * 16;

    // A fragment (16x32): row = mBase+l16, chunks k = hi*8 + {0..7}, +16
    const float* aRow = input + (size_t)(mBase + l16) * INF + hi * 8;
    // B fragment (32x16): col o = oBase+l16, chunks k = hi*8 + {0..7}, +16
    const float* bRow = W + (size_t)(oBase + l16) * INF + hi * 8;

    f32x8 acc = {};
    for (int kb = 0; kb < INF; kb += 32) {
        f32x4 a0 = *(const f32x4*)(aRow + kb);
        f32x4 a1 = *(const f32x4*)(aRow + kb + 4);
        f32x4 a2 = *(const f32x4*)(aRow + kb + 16);
        f32x4 a3 = *(const f32x4*)(aRow + kb + 20);
        f32x4 b0 = *(const f32x4*)(bRow + kb);
        f32x4 b1 = *(const f32x4*)(bRow + kb + 4);
        f32x4 b2 = *(const f32x4*)(bRow + kb + 16);
        f32x4 b3 = *(const f32x4*)(bRow + kb + 20);

        bf16x16 av, bv;
        #pragma unroll
        for (int i = 0; i < 4; ++i) {
            av[i]      = (bf16_t)a0[i];
            av[4 + i]  = (bf16_t)a1[i];
            av[8 + i]  = (bf16_t)a2[i];
            av[12 + i] = (bf16_t)a3[i];
            bv[i]      = (bf16_t)b0[i];
            bv[4 + i]  = (bf16_t)b1[i];
            bv[8 + i]  = (bf16_t)b2[i];
            bv[12 + i] = (bf16_t)b3[i];
        }
        acc = __builtin_amdgcn_wmma_f32_16x16x32_bf16(
            false, av, false, bv, (short)0, acc, false, false);
    }

    const float bb = bias[oBase + l16];
    bf16_t* outCol = hT + (size_t)(oBase + l16) * NROWS + mBase + hi * 8;
    #pragma unroll
    for (int v = 0; v < 8; ++v)
        outCol[v] = (bf16_t)(acc[v] + bb);
}

// ---------------------------------------------------------------------------
// Kernel 2: out[m][o] = sum_n adj[m][n] * h[n][o]
// Workgroup: 64 rows x full 512 cols -> adj streamed from HBM exactly once.
// Double-buffered LDS, one barrier per K-step:
//   - B tile (512x32 bf16 of hT) DMA'd by the Tensor Data Mover (wave 0),
//     padded rows (80B pitch), completion via s_wait_tensorcnt.
//   - A tile (64x32 fp32 of adj) loaded/converted/stored by all threads,
//     overlapped with the 16-WMMA compute burst on the current buffer.
// ---------------------------------------------------------------------------
__global__ __launch_bounds__(256)
void aggregate_bf16_kernel(const float*  __restrict__ adj, // [NROWS][NROWS]
                           const bf16_t* __restrict__ hT,  // [OUTF][NROWS]
                           float* __restrict__ out)        // [NROWS][OUTF]
{
    __shared__ bf16_t ldsA[2][64][40];    // 10 KB  (40-elem pitch: bank-friendly)
    __shared__ bf16_t ldsB[2][512][40];   // 80 KB  (pitch set by TDM padding)

    const int wave = threadIdx.x >> 5;
    const int lane = threadIdx.x & 31;
    const int l16  = lane & 15;
    const int hi   = lane >> 4;
    const int wm   = wave & 1;            // 0..1 -> 32-row group
    const int wn   = wave >> 1;           // 0..3 -> 128-col group
    const int blockM = blockIdx.x * 64;

    // adj staging: 512 float4 slots (64 rows x 8), thread t does slot t, t+256
    const int r0 = threadIdx.x >> 3;      // 0..31
    const int c4 = threadIdx.x & 7;
    const float* aSrc0 = adj + (size_t)(blockM + r0)      * NROWS + c4 * 4;
    const float* aSrc1 = adj + (size_t)(blockM + r0 + 32) * NROWS + c4 * 4;

    f32x8 acc[2][8] = {};

    // ---------------- prologue: fill buffer 0 ----------------
    if (wave == 0)
        tdm_load_b_tile((unsigned)(unsigned long long)&ldsB[0][0][0], hT);
    {
        f32x4 v0 = *(const f32x4*)(aSrc0);
        f32x4 v1 = *(const f32x4*)(aSrc1);
        bf16x4 w0, w1;
        #pragma unroll
        for (int i = 0; i < 4; ++i) { w0[i] = (bf16_t)v0[i]; w1[i] = (bf16_t)v1[i]; }
        *(bf16x4*)&ldsA[0][r0][c4 * 4]      = w0;
        *(bf16x4*)&ldsA[0][r0 + 32][c4 * 4] = w1;
    }
    if (wave == 0) __builtin_amdgcn_s_wait_tensorcnt(0);
    __syncthreads();

    for (int kb = 0; kb < NROWS; kb += 32) {
        const int  cur  = (kb >> 5) & 1;
        const int  nxt  = cur ^ 1;
        const bool more = (kb + 32) < NROWS;

        // ---- kick off next K-slice: TDM for B, global loads for A ----
        f32x4 v0 = {}, v1 = {};
        if (more) {
            if (wave == 0)
                tdm_load_b_tile((unsigned)(unsigned long long)&ldsB[nxt][0][0],
                                hT + kb + 32);
            v0 = *(const f32x4*)(aSrc0 + kb + 32);
            v1 = *(const f32x4*)(aSrc1 + kb + 32);
        }

        // ---- compute on current buffers: 2 A-frags, 8 B-frags, 16 WMMAs ----
        bf16x16 af[2];
        #pragma unroll
        for (int am = 0; am < 2; ++am) {
            const bf16_t* ar = &ldsA[cur][wm * 32 + am * 16 + l16][hi * 8];
            bf16x8 c0 = *(const bf16x8*)ar;          // k = hi*8 + 0..7
            bf16x8 c1 = *(const bf16x8*)(ar + 16);   // k = 16 + hi*8 + 0..7
            #pragma unroll
            for (int i = 0; i < 8; ++i) { af[am][i] = c0[i]; af[am][8 + i] = c1[i]; }
        }
        #pragma unroll
        for (int ot = 0; ot < 8; ++ot) {
            const bf16_t* bp = &ldsB[cur][wn * 128 + ot * 16 + l16][hi * 8];
            bf16x8 b0 = *(const bf16x8*)bp;
            bf16x8 b1 = *(const bf16x8*)(bp + 16);
            bf16x16 bv;
            #pragma unroll
            for (int i = 0; i < 8; ++i) { bv[i] = b0[i]; bv[8 + i] = b1[i]; }

            acc[0][ot] = __builtin_amdgcn_wmma_f32_16x16x32_bf16(
                false, af[0], false, bv, (short)0, acc[0][ot], false, false);
            acc[1][ot] = __builtin_amdgcn_wmma_f32_16x16x32_bf16(
                false, af[1], false, bv, (short)0, acc[1][ot], false, false);
        }

        // ---- finish staging next A tile into the other buffer ----
        if (more) {
            bf16x4 w0, w1;
            #pragma unroll
            for (int i = 0; i < 4; ++i) { w0[i] = (bf16_t)v0[i]; w1[i] = (bf16_t)v1[i]; }
            *(bf16x4*)&ldsA[nxt][r0][c4 * 4]      = w0;
            *(bf16x4*)&ldsA[nxt][r0 + 32][c4 * 4] = w1;
        }
        if (wave == 0) __builtin_amdgcn_s_wait_tensorcnt(0);
        __syncthreads();
    }

    // ---- writeback: D lane col = l16, VGPR v -> row = v (+8 for hi lanes) ----
    #pragma unroll
    for (int am = 0; am < 2; ++am) {
        #pragma unroll
        for (int ot = 0; ot < 8; ++ot) {
            const int col  = wn * 128 + ot * 16 + l16;
            const int row0 = blockM + wm * 32 + am * 16 + hi * 8;
            #pragma unroll
            for (int v = 0; v < 8; ++v)
                out[(size_t)(row0 + v) * OUTF + col] = acc[am][ot][v];
        }
    }
}

extern "C" void kernel_launch(void* const* d_in, const int* in_sizes, int n_in,
                              void* d_out, int out_size, void* d_ws, size_t ws_size,
                              hipStream_t stream) {
    const float* input = (const float*)d_in[0];  // [8192, 512]
    const float* adj   = (const float*)d_in[1];  // [8192, 8192]
    const float* W     = (const float*)d_in[2];  // [512, 512]
    const float* b     = (const float*)d_in[3];  // [512]
    bf16_t* hT = (bf16_t*)d_ws;                  // [512][8192] bf16 = 8 MB

    linear_bf16_kernel<<<2048, 256, 0, stream>>>(input, W, b, hT);
    aggregate_bf16_kernel<<<128, 256, 0, stream>>>(adj, hT, (float*)d_out);
}
